// T5LayerFF_Combined_54846732370206
// MI455X (gfx1250) — matrix operand order repaired and checked by reading, not compiled, not run on hardware
//
#include <hip/hip_runtime.h>
#include <hip/hip_bf16.h>

// ---------------------------------------------------------------------------
// T5 MoE FF layer for MI455X (gfx1250, wave32, WMMA).
// B=4, S=2048 -> T=8192 tokens, D=1024, FF=4096, E=16, top-2 routing.
// ---------------------------------------------------------------------------

typedef __bf16 bf16_t;
typedef __attribute__((ext_vector_type(16))) __bf16 v16bf;
typedef __attribute__((ext_vector_type(8)))  __bf16 v8bf;
typedef __attribute__((ext_vector_type(2)))  __bf16 v2bf;
typedef __attribute__((ext_vector_type(8)))  float  v8f;

union Frag16 { v16bf v; v8bf h[2]; };

constexpr int kD   = 1024;
constexpr int kFF  = 4096;
constexpr int kE   = 16;
constexpr int kT   = 8192;          // B*S tokens
constexpr int TILE_M = 64;          // tokens per block
constexpr int TILE_F = 128;         // FF chunk
constexpr int KSTEP  = 32;          // WMMA K per step (bf16)
constexpr float kEPS = 1e-6f;

// ---------------------------------------------------------------------------
// Kernel 1: RMS norm + gate (top-2 softmax) + routing lists. One wave / token.
// ---------------------------------------------------------------------------
__global__ __launch_bounds__(256)
void gate_norm_kernel(const float* __restrict__ x,
                      const float* __restrict__ lnw,
                      const float* __restrict__ gw,   // [D, E]
                      const float* __restrict__ gb,   // [E]
                      bf16_t* __restrict__ xn,        // [T, D] bf16
                      int* __restrict__ counts,       // [E]
                      int* __restrict__ ltok,         // [E, T]
                      float* __restrict__ lw)         // [E, T]
{
    const int lane = threadIdx.x & 31;
    const int wv   = threadIdx.x >> 5;
    const int t    = blockIdx.x * 8 + wv;
    if (t >= kT) return;

    const float* xr = x + (size_t)t * kD;
    float xv[32];
    float ss = 0.f;
    #pragma unroll
    for (int j = 0; j < 32; ++j) {
        float v = xr[lane + 32 * j];
        xv[j] = v;
        ss += v * v;
    }
    #pragma unroll
    for (int off = 16; off; off >>= 1) ss += __shfl_xor(ss, off, 32);
    const float rinv = rsqrtf(ss * (1.0f / kD) + kEPS);

    float acc[kE];
    #pragma unroll
    for (int e = 0; e < kE; ++e) acc[e] = 0.f;

    bf16_t* xnr = xn + (size_t)t * kD;
    #pragma unroll
    for (int j = 0; j < 32; ++j) {
        const int d = lane + 32 * j;
        const float nv = xv[j] * rinv * lnw[d];
        xnr[d] = (bf16_t)nv;
        const float* g = gw + (size_t)d * kE;
        #pragma unroll
        for (int e = 0; e < kE; ++e) acc[e] += nv * g[e];
    }
    #pragma unroll
    for (int e = 0; e < kE; ++e) {
        #pragma unroll
        for (int off = 16; off; off >>= 1) acc[e] += __shfl_xor(acc[e], off, 32);
    }

    if (lane == 0) {
        float logit[kE];
        float v0 = -1e30f; int i0 = 0;
        #pragma unroll
        for (int e = 0; e < kE; ++e) {
            float l = acc[e] + gb[e];
            logit[e] = l;
            if (l > v0) { v0 = l; i0 = e; }
        }
        float v1 = -1e30f; int i1 = 0;
        #pragma unroll
        for (int e = 0; e < kE; ++e) {
            if (e != i0 && logit[e] > v1) { v1 = logit[e]; i1 = e; }
        }
        // softmax over the two selected logits
        const float e1 = __expf(v1 - v0);
        const float inv = 1.0f / (1.0f + e1);
        const float w0 = inv;
        const float w1 = e1 * inv;

        int p0 = atomicAdd(&counts[i0], 1);
        ltok[i0 * kT + p0] = t;
        lw  [i0 * kT + p0] = w0;
        int p1 = atomicAdd(&counts[i1], 1);
        ltok[i1 * kT + p1] = t;
        lw  [i1 * kT + p1] = w1;
    }
}

// ---------------------------------------------------------------------------
// Kernel 2: routed expert FFN.  Block = (expert, 64-token tile), 8 waves.
//  GEMM1: H = relu(X[64,1024] @ wi[1024, fchunk:128])  via wmma bf16
//  GEMM2: Y += H[64,128] @ wo[fchunk:128, 1024]        via wmma bf16
//  Y scaled by gate weight, atomically added into residual-initialized out.
//  Weight staging converts fp32->bf16 with PACKED cvt + ds_store_b32:
//  LDS layout is [col][k] with K contiguous, so K-row pairs pack naturally.
// ---------------------------------------------------------------------------
__global__ __launch_bounds__(256)
void moe_ffn_kernel(const bf16_t* __restrict__ xn,    // [T, D] bf16
                    const float* __restrict__ wi,     // [E, D, FF]
                    const float* __restrict__ wo,     // [E, FF, D]
                    const int* __restrict__ counts,
                    const int* __restrict__ ltok,
                    const float* __restrict__ lw,
                    float* __restrict__ out)          // [T, D], holds residual
{
    extern __shared__ char smem[];
    bf16_t* sH   = (bf16_t*)(smem);                 // 64*128*2  = 16 KB
    bf16_t* sB1  = (bf16_t*)(smem + 16 * 1024);     // 128*32*2  =  8 KB
    bf16_t* sB2  = (bf16_t*)(smem + 24 * 1024);     // 1024*32*2 = 64 KB
    int*    sTok = (int*)   (smem + 88 * 1024);     // 64 * 4
    float*  sW   = (float*) (smem + 88 * 1024 + 256);

    const int e    = blockIdx.x;
    const int tile = blockIdx.y;
    const int cnt  = counts[e];
    if (tile * TILE_M >= cnt) return;

    const int tid = threadIdx.x;
    if (tid < TILE_M) {
        const int r = tile * TILE_M + tid;
        sTok[tid] = (r < cnt) ? ltok[e * kT + r] : 0;
        sW  [tid] = (r < cnt) ? lw  [e * kT + r] : 0.f;
    }
    __syncthreads();

    const int lane    = tid & 31;
    const int wv      = tid >> 5;        // 0..7
    const int l15     = lane & 15;
    const int halfsel = lane >> 4;       // 0 or 1
    const int alo = halfsel * 8;         // A-frag K offsets {0,+16} / {8,+16}
    const int klo = halfsel * 16;        // unused; B-frags also load with alo

    // staging decomposition: each thread owns K-row pair {2*rp, 2*rp+1}
    const int rp  = tid >> 4;            // 0..15
    const int cg  = tid & 15;            // column group

    // Per-lane A row base pointers (gathered token rows), 4 M-tiles of 16.
    const bf16_t* aRow[4];
    #pragma unroll
    for (int mt = 0; mt < 4; ++mt)
        aRow[mt] = xn + (size_t)sTok[mt * 16 + l15] * kD;

    const float* Wi = wi + (size_t)e * kD * kFF;
    const float* Wo = wo + (size_t)e * kFF * kD;

    const v8f vzero = {0.f, 0.f, 0.f, 0.f, 0.f, 0.f, 0.f, 0.f};
    v8f yacc[4][8];
    #pragma unroll
    for (int mt = 0; mt < 4; ++mt)
        #pragma unroll
        for (int nt = 0; nt < 8; ++nt) yacc[mt][nt] = vzero;

    for (int fc = 0; fc < kFF; fc += TILE_F) {
        // ---------------- GEMM1: H chunk, wave owns 16 FF columns ----------
        v8f hacc[4];
        #pragma unroll
        for (int mt = 0; mt < 4; ++mt) hacc[mt] = vzero;

        for (int k0 = 0; k0 < kD; k0 += KSTEP) {
            __syncthreads();
            {   // stage wi rows [k0,k0+32), cols [fc,fc+128) -> sB1[c*32+k]
                // thread: K rows {k0+2rp, k0+2rp+1}, cols [cg*8, cg*8+8)
                const int c0 = cg * 8;
                const float4* s0 =
                    (const float4*)(Wi + (size_t)(k0 + 2 * rp) * kFF + fc + c0);
                const float4* s1 =
                    (const float4*)(Wi + (size_t)(k0 + 2 * rp + 1) * kFF + fc + c0);
                #pragma unroll
                for (int c4 = 0; c4 < 2; ++c4) {
                    float4 a = s0[c4];
                    float4 b = s1[c4];
                    const int c = c0 + c4 * 4;
                    *(v2bf*)(sB1 + (c + 0) * KSTEP + 2 * rp) =
                        v2bf{(bf16_t)a.x, (bf16_t)b.x};
                    *(v2bf*)(sB1 + (c + 1) * KSTEP + 2 * rp) =
                        v2bf{(bf16_t)a.y, (bf16_t)b.y};
                    *(v2bf*)(sB1 + (c + 2) * KSTEP + 2 * rp) =
                        v2bf{(bf16_t)a.z, (bf16_t)b.z};
                    *(v2bf*)(sB1 + (c + 3) * KSTEP + 2 * rp) =
                        v2bf{(bf16_t)a.w, (bf16_t)b.w};
                }
                if (k0 + KSTEP < kD)
                    __builtin_prefetch(
                        Wi + (size_t)(k0 + KSTEP + 2 * rp) * kFF + fc + c0, 0, 1);
            }
            __syncthreads();

            Frag16 bfB;
            {
                const bf16_t* bp = sB1 + (wv * 16 + l15) * KSTEP + alo;
                bfB.h[0] = *(const v8bf*)(bp);
                bfB.h[1] = *(const v8bf*)(bp + 16);
            }
            #pragma unroll
            for (int mt = 0; mt < 4; ++mt) {
                Frag16 afA;
                const bf16_t* ap = aRow[mt] + k0 + alo;
                afA.h[0] = *(const v8bf*)(ap);
                afA.h[1] = *(const v8bf*)(ap + 16);
                hacc[mt] = __builtin_amdgcn_wmma_f32_16x16x32_bf16(
                    false, afA.v, false, bfB.v, (short)0, hacc[mt], false, false);
            }
        }

        // relu + cvt, wave stores its 16 columns of H
        __syncthreads();
        #pragma unroll
        for (int mt = 0; mt < 4; ++mt) {
            const int col = wv * 16 + l15;
            #pragma unroll
            for (int r = 0; r < 8; ++r) {
                const int row = mt * 16 + r + halfsel * 8;
                float v = hacc[mt][r];
                sH[row * TILE_F + col] = (bf16_t)(v > 0.f ? v : 0.f);
            }
        }
        __syncthreads();

        // ---------------- GEMM2: Y += H @ wo, wave owns 128 D columns ------
        for (int k2 = 0; k2 < TILE_F; k2 += KSTEP) {
            __syncthreads();
            {   // stage wo rows [fc+k2, +32), all 1024 cols -> sB2[n*32+k]
                // thread: K rows {2rp, 2rp+1}, cols [cg*64, cg*64+64)
                const int c0 = cg * 64;
                const float4* s0 =
                    (const float4*)(Wo + (size_t)(fc + k2 + 2 * rp) * kD + c0);
                const float4* s1 =
                    (const float4*)(Wo + (size_t)(fc + k2 + 2 * rp + 1) * kD + c0);
                #pragma unroll
                for (int c4 = 0; c4 < 16; ++c4) {
                    float4 a = s0[c4];
                    float4 b = s1[c4];
                    const int c = c0 + c4 * 4;
                    *(v2bf*)(sB2 + (c + 0) * KSTEP + 2 * rp) =
                        v2bf{(bf16_t)a.x, (bf16_t)b.x};
                    *(v2bf*)(sB2 + (c + 1) * KSTEP + 2 * rp) =
                        v2bf{(bf16_t)a.y, (bf16_t)b.y};
                    *(v2bf*)(sB2 + (c + 2) * KSTEP + 2 * rp) =
                        v2bf{(bf16_t)a.z, (bf16_t)b.z};
                    *(v2bf*)(sB2 + (c + 3) * KSTEP + 2 * rp) =
                        v2bf{(bf16_t)a.w, (bf16_t)b.w};
                }
            }
            __syncthreads();

            Frag16 afA[4];
            #pragma unroll
            for (int mt = 0; mt < 4; ++mt) {
                const bf16_t* ap = sH + (mt * 16 + l15) * TILE_F + k2 + alo;
                afA[mt].h[0] = *(const v8bf*)(ap);
                afA[mt].h[1] = *(const v8bf*)(ap + 16);
            }
            #pragma unroll
            for (int nt = 0; nt < 8; ++nt) {
                Frag16 bfB;
                const bf16_t* bp = sB2 + (wv * 128 + nt * 16 + l15) * KSTEP + alo;
                bfB.h[0] = *(const v8bf*)(bp);
                bfB.h[1] = *(const v8bf*)(bp + 16);
                #pragma unroll
                for (int mt = 0; mt < 4; ++mt) {
                    yacc[mt][nt] = __builtin_amdgcn_wmma_f32_16x16x32_bf16(
                        false, afA[mt].v, false, bfB.v, (short)0, yacc[mt][nt],
                        false, false);
                }
            }
        }
    }

    __syncthreads();
    // ---- scale by gate weight and accumulate into residual-initialized out
    #pragma unroll
    for (int mt = 0; mt < 4; ++mt) {
        #pragma unroll
        for (int r = 0; r < 8; ++r) {
            const int row = mt * 16 + r + halfsel * 8;
            const int tok = sTok[row];
            const float cw = sW[row];
            float* base = out + (size_t)tok * kD + wv * 128 + l15;
            #pragma unroll
            for (int nt = 0; nt < 8; ++nt) {
                __hip_atomic_fetch_add(base + nt * 16, yacc[mt][nt][r] * cw,
                                       __ATOMIC_RELAXED,
                                       __HIP_MEMORY_SCOPE_AGENT);
            }
        }
    }
}

// ---------------------------------------------------------------------------
extern "C" void kernel_launch(void* const* d_in, const int* in_sizes, int n_in,
                              void* d_out, int out_size, void* d_ws, size_t ws_size,
                              hipStream_t stream) {
    const float* x   = (const float*)d_in[0];   // hidden_states [B,S,D]
    const float* lnw = (const float*)d_in[1];   // ln_weight [D]
    const float* gw  = (const float*)d_in[2];   // gate_w [D,E]
    const float* gb  = (const float*)d_in[3];   // gate_b [E]
    const float* wi  = (const float*)d_in[4];   // [E,D,FF]
    const float* wo  = (const float*)d_in[5];   // [E,FF,D]
    float* out = (float*)d_out;

    char* ws = (char*)d_ws;
    bf16_t* xn = (bf16_t*)ws;                               // 16.78 MB
    size_t off = (size_t)kT * kD * sizeof(bf16_t);
    int* counts = (int*)(ws + off);  off += 256;
    int* ltokL  = (int*)(ws + off);  off += (size_t)kE * kT * sizeof(int);
    float* lwL  = (float*)(ws + off);

    // residual init of output + zero routing counts (capture-safe async ops)
    hipMemcpyAsync(out, x, (size_t)kT * kD * sizeof(float),
                   hipMemcpyDeviceToDevice, stream);
    hipMemsetAsync(counts, 0, 256, stream);

    gate_norm_kernel<<<kT / 8, 256, 0, stream>>>(x, lnw, gw, gb, xn,
                                                 counts, ltokL, lwL);

    dim3 grid(kE, kT / TILE_M);                  // (16, 128); extra tiles exit
    const size_t shmem = 88 * 1024 + 512;        // sH+sB1+sB2+token/weight
    moe_ffn_kernel<<<grid, 256, shmem, stream>>>(xn, wi, wo, counts,
                                                 ltokL, lwL, out);
}
